// Attention_2826088481501
// MI455X (gfx1250) — compile-verified, matrix operand order checked by reading, not run on hardware
//
#include <hip/hip_runtime.h>
#include <hip/hip_bf16.h>

#define B_ 2
#define T_ 2048
#define C_ 2048
#define H_ 16
#define KV_ 4
#define D_ 128
#define G_ (H_ / KV_)

typedef __attribute__((ext_vector_type(16))) __bf16 v16bf;
typedef __attribute__((ext_vector_type(8)))  float  v8f;

__device__ __forceinline__ __bf16 f2bf(float f) {
    unsigned u = __float_as_uint(f);
    unsigned r = (u + 0x7FFFu + ((u >> 16) & 1u)) >> 16;
    unsigned short h = (unsigned short)r;
    __bf16 b;
    __builtin_memcpy(&b, &h, 2);
    return b;
}

__device__ __forceinline__ v8f zero8() {
    v8f z = {0.f, 0.f, 0.f, 0.f, 0.f, 0.f, 0.f, 0.f};
    return z;
}

__device__ __forceinline__ v8f wmma_bf16(v16bf a, v16bf b, v8f c) {
    return __builtin_amdgcn_wmma_f32_16x16x32_bf16(false, a, false, b, (short)0, c, false, false);
}

// 16x32 bf16 fragment (A-layout, also used for B with "row" == output column).
// Lane l: row = l&15; holds K = { (l>>4)*8 + 0..7 , 16 + (l>>4)*8 + 0..7 }.
// Any consistent K mapping shared by A and B fragments yields a correct dot.
__device__ __forceinline__ v16bf load_frag(const __bf16* p0, int stride) {
    const int lane = threadIdx.x & 31;
    const __bf16* p = p0 + (size_t)(lane & 15) * (size_t)stride + ((lane >> 4) << 3);
    union { v16bf v; uint4 q[2]; } u;
    u.q[0] = *(const uint4*)p;
    u.q[1] = *(const uint4*)(p + 16);
    return u.v;
}

__device__ __forceinline__ v16bf load_frag_lds(const __bf16 (&t)[16][32]) {
    const int lane = threadIdx.x & 31;
    const int row  = lane & 15;
    const int ko   = (lane >> 4) << 3;
    union { v16bf v; uint4 q[2]; } u;
    u.q[0] = *(const uint4*)&t[row][ko];
    u.q[1] = *(const uint4*)&t[row][ko + 16];
    return u.v;
}

// f32 16x16 C/D layout: VGPR r -> row r (lanes 0-15) / r+8 (lanes 16-31), col = lane&15
__device__ __forceinline__ void store_tile_f32(float* p, int stride, v8f d) {
    const int lane = threadIdx.x & 31;
    const int col  = lane & 15;
    const int rb   = (lane >> 4) << 3;
#pragma unroll
    for (int r = 0; r < 8; ++r)
        p[(size_t)(r + rb) * (size_t)stride + col] = d[r];
}

// ---------------- elementwise kernels ----------------

__global__ void cast_f32_bf16(const float* __restrict__ in, __bf16* __restrict__ out, long n) {
    long i = (long)blockIdx.x * blockDim.x + threadIdx.x;
    if (i < n) out[i] = f2bf(in[i]);
}

__global__ void rope_q(const float* __restrict__ Qf, const float* __restrict__ cv,
                       const float* __restrict__ sv, __bf16* __restrict__ Qb) {
    long i = (long)blockIdx.x * blockDim.x + threadIdx.x;
    if (i >= (long)B_ * T_ * H_ * (D_ / 2)) return;
    int d2 = (int)(i % (D_ / 2)); long r = i / (D_ / 2);
    int h = (int)(r % H_); r /= H_;
    int t = (int)(r % T_); int b = (int)(r / T_);
    size_t base = (((size_t)(b * T_ + t)) * H_ + h) * D_ + 2 * d2;
    float c = cv[t * (D_ / 2) + d2], s = sv[t * (D_ / 2) + d2];
    float q0 = Qf[base], q1 = Qf[base + 1];
    Qb[base]     = f2bf(q0 * c - q1 * s);
    Qb[base + 1] = f2bf(q0 * s + q1 * c);
}

__global__ void rope_k(const float* __restrict__ Kf, const float* __restrict__ cv,
                       const float* __restrict__ sv, __bf16* __restrict__ Kb) {
    long i = (long)blockIdx.x * blockDim.x + threadIdx.x;
    if (i >= (long)B_ * T_ * KV_ * (D_ / 2)) return;
    int d2 = (int)(i % (D_ / 2)); long r = i / (D_ / 2);
    int kv = (int)(r % KV_); r /= KV_;
    int t = (int)(r % T_); int b = (int)(r / T_);
    size_t base = (((size_t)(b * T_ + t)) * KV_ + kv) * D_ + 2 * d2;
    float c = cv[t * (D_ / 2) + d2], s = sv[t * (D_ / 2) + d2];
    float k0 = Kf[base], k1 = Kf[base + 1];
    Kb[base]     = f2bf(k0 * c - k1 * s);
    Kb[base + 1] = f2bf(k0 * s + k1 * c);
}

// V [b][t][kv][d] (f32) -> Vt [b][kv][d][t] (bf16), so PV B-fragments load contiguously in t
__global__ void transpose_v(const float* __restrict__ Vf, __bf16* __restrict__ Vt) {
    long i = (long)blockIdx.x * blockDim.x + threadIdx.x;
    if (i >= (long)B_ * KV_ * D_ * T_) return;
    int t = (int)(i % T_); long r = i / T_;
    int d = (int)(r % D_); r /= D_;
    int kv = (int)(r % KV_); int b = (int)(r / KV_);
    Vt[i] = f2bf(Vf[(((size_t)(b * T_ + t)) * KV_ + kv) * D_ + d]);
}

// ---------------- WMMA GEMM: Y[M,N] = X[M,K] * W[N,K]^T ----------------
// 256 threads = 8 waves; wave -> 32x64 tile (2 A-frags x 4 B-frags, 8 WMMA / 6 loads
// per K=32 step -> 24 B/lane/WMMA); block tile 64x256 (waves 2x4).
__global__ void __launch_bounds__(256) gemm_xt(const __bf16* __restrict__ X,
                                               const __bf16* __restrict__ W,
                                               float* __restrict__ Y,
                                               int M, int N, int K) {
    const int wave = threadIdx.x >> 5;
    const int m0 = blockIdx.x * 64 + (wave >> 2) * 32;
    const int n0 = blockIdx.y * 256 + (wave & 3) * 64;
    v8f acc[2][4];
#pragma unroll
    for (int i = 0; i < 2; ++i)
#pragma unroll
        for (int j = 0; j < 4; ++j) acc[i][j] = zero8();

    const __bf16* Xp = X + (size_t)m0 * K;
    const __bf16* Wp = W + (size_t)n0 * K;
    for (int k = 0; k < K; k += 32) {
        // speculative prefetch ~8 K-steps ahead (dropped silently if OOB)
        __builtin_prefetch(Xp + k + 256, 0, 0);
        __builtin_prefetch(Wp + k + 256, 0, 0);
        v16bf fa0 = load_frag(Xp + k, K);
        v16bf fa1 = load_frag(Xp + (size_t)16 * K + k, K);
#pragma unroll
        for (int j = 0; j < 4; ++j) {
            v16bf fb = load_frag(Wp + (size_t)(j * 16) * K + k, K);
            acc[0][j] = wmma_bf16(fa0, fb, acc[0][j]);
            acc[1][j] = wmma_bf16(fa1, fb, acc[1][j]);
        }
    }
    float* Y0 = Y + (size_t)m0 * N + n0;
#pragma unroll
    for (int i = 0; i < 2; ++i)
#pragma unroll
        for (int j = 0; j < 4; ++j)
            store_tile_f32(Y0 + (size_t)(i * 16) * N + j * 16, N, acc[i][j]);
}

// ---------------- Flash attention ----------------
// grid.x = B*H*(T/128); block = 256 (8 waves); wave w handles q rows [qc*128+w*16, +16)
__global__ void __launch_bounds__(256) attn_fa(const __bf16* __restrict__ Qb,
                                               const __bf16* __restrict__ Kb,
                                               const __bf16* __restrict__ Vt,
                                               __bf16* __restrict__ Ob) {
    __shared__ __align__(16) __bf16 plds[8][16][32];
    const int wave = threadIdx.x >> 5;
    const int lane = threadIdx.x & 31;
    const int hl = lane >> 4;   // half-wave id (row-group select)
    const int cl = lane & 15;   // column within tile
    const int qchunks = T_ / 128;
    const int bh = blockIdx.x / qchunks;
    const int qc = blockIdx.x % qchunks;
    const int b = bh / H_, h = bh % H_, kvh = h / G_;
    const int qg0 = qc * 128 + wave * 16;
    const float scale = 0.08838834764831845f; // 1/sqrt(128)

    v16bf qf[4];
    const __bf16* Qp = Qb + (((size_t)(b * T_ + qg0)) * H_ + h) * D_;
#pragma unroll
    for (int j = 0; j < 4; ++j) qf[j] = load_frag(Qp + j * 32, H_ * D_);

    float mrow[8], lrow[8];
    v8f o[8];
#pragma unroll
    for (int r = 0; r < 8; ++r) { mrow[r] = -3.0e38f; lrow[r] = 0.f; }
#pragma unroll
    for (int j = 0; j < 8; ++j) o[j] = zero8();

    const int kend = qg0 + 16; // exclusive causal key limit
    const __bf16* Kbase = Kb + ((size_t)b * T_ * KV_ + kvh) * D_;
    const __bf16* Vbase = Vt + ((size_t)(b * KV_ + kvh)) * D_ * T_;

    for (int k0 = 0; k0 < kend; k0 += 32) {
        v8f s0 = zero8(), s1 = zero8();
#pragma unroll
        for (int j = 0; j < 4; ++j) {
            v16bf kb0 = load_frag(Kbase + (size_t)k0 * KV_ * D_ + j * 32, KV_ * D_);
            v16bf kb1 = load_frag(Kbase + (size_t)(k0 + 16) * KV_ * D_ + j * 32, KV_ * D_);
            s0 = wmma_bf16(qf[j], kb0, s0);
            s1 = wmma_bf16(qf[j], kb1, s1);
        }
        // scale + causal mask (set, don't add, so no garbage can reach exp)
        float p0[8], p1[8], mt[8];
#pragma unroll
        for (int r = 0; r < 8; ++r) {
            int qrow = qg0 + r + hl * 8;
            float v0 = (k0 + cl <= qrow)      ? s0[r] * scale : -3.0e38f;
            float v1 = (k0 + 16 + cl <= qrow) ? s1[r] * scale : -3.0e38f;
            p0[r] = v0; p1[r] = v1;
            mt[r] = fmaxf(v0, v1);
        }
        // row max across the 16 lanes of each half-wave
#pragma unroll
        for (int x = 1; x < 16; x <<= 1) {
#pragma unroll
            for (int r = 0; r < 8; ++r) mt[r] = fmaxf(mt[r], __shfl_xor(mt[r], x, 32));
        }
        float rs[8];
#pragma unroll
        for (int r = 0; r < 8; ++r) {
            float mnew  = fmaxf(mrow[r], mt[r]);
            float alpha = __expf(mrow[r] - mnew);
            p0[r] = __expf(p0[r] - mnew);
            p1[r] = __expf(p1[r] - mnew);
            rs[r] = p0[r] + p1[r];
            lrow[r] *= alpha;
            mrow[r] = mnew;
#pragma unroll
            for (int j = 0; j < 8; ++j) o[j][r] *= alpha;
        }
#pragma unroll
        for (int x = 1; x < 16; x <<= 1) {
#pragma unroll
            for (int r = 0; r < 8; ++r) rs[r] += __shfl_xor(rs[r], x, 32);
        }
#pragma unroll
        for (int r = 0; r < 8; ++r) lrow[r] += rs[r];

        // P: D-layout -> row-major LDS (16x32 bf16) -> A-fragment
#pragma unroll
        for (int r = 0; r < 8; ++r) {
            int row = r + hl * 8;
            plds[wave][row][cl]      = f2bf(p0[r]);
            plds[wave][row][cl + 16] = f2bf(p1[r]);
        }
        __builtin_amdgcn_wave_barrier();
        asm volatile("s_wait_dscnt 0" ::: "memory");
        v16bf pf = load_frag_lds(plds[wave]);
        __builtin_amdgcn_wave_barrier();
#pragma unroll
        for (int j = 0; j < 8; ++j) {
            v16bf vb = load_frag(Vbase + (size_t)(j * 16) * T_ + k0, T_);
            o[j] = wmma_bf16(pf, vb, o[j]);
        }
    }

    float inv[8];
#pragma unroll
    for (int r = 0; r < 8; ++r) inv[r] = 1.f / lrow[r];
    __bf16* Op = Ob + (((size_t)(b * T_ + qg0)) * H_ + h) * D_;
#pragma unroll
    for (int j = 0; j < 8; ++j)
#pragma unroll
        for (int r = 0; r < 8; ++r)
            Op[(size_t)(r + hl * 8) * (H_ * D_) + j * 16 + cl] = f2bf(o[j][r] * inv[r]);
}

// ---------------- host launcher ----------------

extern "C" void kernel_launch(void* const* d_in, const int* in_sizes, int n_in,
                              void* d_out, int out_size, void* d_ws, size_t ws_size,
                              hipStream_t stream) {
    (void)in_sizes; (void)n_in; (void)out_size; (void)ws_size;
    const float* x  = (const float*)d_in[0];
    const float* wq = (const float*)d_in[1];
    const float* wk = (const float*)d_in[2];
    const float* wv = (const float*)d_in[3];
    const float* wo = (const float*)d_in[4];
    const float* fc = (const float*)d_in[5];
    const float* fs = (const float*)d_in[6];
    // d_in[7] (mask) unused: causal mask applied analytically.

    char* w = (char*)d_ws;
    size_t off = 0;
    auto carve = [&](size_t bytes) -> char* {
        char* p = w + off;
        off += (bytes + 255) & ~(size_t)255;
        return p;
    };
    const size_t MT = (size_t)B_ * T_;             // 4096 rows
    __bf16* xb  = (__bf16*)carve(MT * C_ * 2);
    __bf16* wqb = (__bf16*)carve((size_t)H_ * D_ * C_ * 2);
    __bf16* wkb = (__bf16*)carve((size_t)KV_ * D_ * C_ * 2);
    __bf16* wvb = (__bf16*)carve((size_t)KV_ * D_ * C_ * 2);
    __bf16* wob = (__bf16*)carve((size_t)C_ * H_ * D_ * 2);
    float*  Qf  = (float*)carve(MT * H_ * D_ * 4);
    float*  Kf  = (float*)carve(MT * KV_ * D_ * 4);
    float*  Vf  = (float*)carve(MT * KV_ * D_ * 4);
    __bf16* Qb  = (__bf16*)carve(MT * H_ * D_ * 2);
    __bf16* Kb  = (__bf16*)carve(MT * KV_ * D_ * 2);
    __bf16* Vt  = (__bf16*)carve(MT * KV_ * D_ * 2);
    __bf16* Ab  = (__bf16*)carve(MT * H_ * D_ * 2);

    auto nblk = [](long n) { return dim3((unsigned)((n + 255) / 256)); };

    cast_f32_bf16<<<nblk((long)MT * C_), 256, 0, stream>>>(x, xb, (long)MT * C_);
    cast_f32_bf16<<<nblk((long)H_ * D_ * C_), 256, 0, stream>>>(wq, wqb, (long)H_ * D_ * C_);
    cast_f32_bf16<<<nblk((long)KV_ * D_ * C_), 256, 0, stream>>>(wk, wkb, (long)KV_ * D_ * C_);
    cast_f32_bf16<<<nblk((long)KV_ * D_ * C_), 256, 0, stream>>>(wv, wvb, (long)KV_ * D_ * C_);
    cast_f32_bf16<<<nblk((long)C_ * H_ * D_), 256, 0, stream>>>(wo, wob, (long)C_ * H_ * D_);

    dim3 blk(256);
    gemm_xt<<<dim3((unsigned)(MT / 64), (H_ * D_) / 256), blk, 0, stream>>>(xb, wqb, Qf, (int)MT, H_ * D_, C_);
    gemm_xt<<<dim3((unsigned)(MT / 64), (KV_ * D_) / 256), blk, 0, stream>>>(xb, wkb, Kf, (int)MT, KV_ * D_, C_);
    gemm_xt<<<dim3((unsigned)(MT / 64), (KV_ * D_) / 256), blk, 0, stream>>>(xb, wvb, Vf, (int)MT, KV_ * D_, C_);

    rope_q<<<nblk((long)B_ * T_ * H_ * (D_ / 2)), 256, 0, stream>>>(Qf, fc, fs, Qb);
    rope_k<<<nblk((long)B_ * T_ * KV_ * (D_ / 2)), 256, 0, stream>>>(Kf, fc, fs, Kb);
    transpose_v<<<nblk((long)B_ * KV_ * D_ * T_), 256, 0, stream>>>(Vf, Vt);

    attn_fa<<<dim3(B_ * H_ * (T_ / 128)), blk, 0, stream>>>(Qb, Kb, Vt, Ab);

    gemm_xt<<<dim3((unsigned)(MT / 64), C_ / 256), blk, 0, stream>>>(Ab, wob, (float*)d_out, (int)MT, C_, H_ * D_);
}
